// CrossAttentionFusion_4054449127538
// MI455X (gfx1250) — matrix operand order repaired and checked by reading, not compiled, not run on hardware
//
#include <hip/hip_runtime.h>

// ---------------------------------------------------------------------------
// CrossAttentionFusion on MI455X (gfx1250, wave32, WMMA bf16 16x16x32 + TDM)
// B=4, C=Cs=256, CI=128, H=W=64 -> N=M=4096
// ---------------------------------------------------------------------------

typedef __attribute__((ext_vector_type(16))) __bf16 bf16x16;
typedef __attribute__((ext_vector_type(8)))  __bf16 bf16x8;
typedef __attribute__((ext_vector_type(8)))  float  f32x8;
typedef __attribute__((ext_vector_type(4)))  float  f32x4;
typedef __attribute__((ext_vector_type(4)))  unsigned int u32x4;
typedef __attribute__((ext_vector_type(8)))  int    i32x8;
typedef __attribute__((ext_vector_type(4)))  int    i32x4;

static constexpr int B_  = 4;
static constexpr int C_  = 256;
static constexpr int Cs_ = 256;
static constexpr int CI_ = 128;
static constexpr int N_  = 4096;   // H*W, also M

static __device__ __forceinline__ f32x8 wmma_bf16(const bf16x16& a, const bf16x16& b, const f32x8& c) {
    return __builtin_amdgcn_wmma_f32_16x16x32_bf16(false, a, false, b, (short)0, c, false, false);
}
static __device__ __forceinline__ bf16x16 cat8(bf16x8 lo, bf16x8 hi) {
    return __builtin_shufflevector(lo, hi, 0,1,2,3,4,5,6,7,8,9,10,11,12,13,14,15);
}

// ---------------------------------------------------------------------------
// Tensor Data Mover: 2-D bf16 tile load, Global -> LDS (ISA 08 D# layout).
// tile_d0 = contiguous elements per row, tile_d1 = rows,
// stride  = row stride in elements; data_size = 2 bytes.
// 6-arg builtin form (clang-23 / therock-10.0 headers).
// ---------------------------------------------------------------------------
static __device__ __forceinline__ void tdm_load_2d_bf16(
    unsigned lds_addr, const void* gptr,
    unsigned tile_d0, unsigned tile_d1,
    unsigned tensor_d0, unsigned tensor_d1, unsigned stride)
{
    const unsigned long long ga = (unsigned long long)(uintptr_t)gptr;
    u32x4 g0;
    g0[0] = 1u;                                              // count=1, user D#
    g0[1] = lds_addr;                                        // lds_addr (bytes)
    g0[2] = (unsigned)(ga & 0xFFFFFFFFu);                    // global_addr[31:0]
    g0[3] = (unsigned)((ga >> 32) & 0x01FFFFFFu) | (2u << 30); // [56:32] + type=2
    i32x8 g1;
    g1[0] = (int)(1u << 16);                                 // data_size=1 -> 2B
    g1[1] = (int)((tensor_d0 & 0xFFFFu) << 16);              // tensor_dim0 lo
    g1[2] = (int)(((tensor_d0 >> 16) & 0xFFFFu) | ((tensor_d1 & 0xFFFFu) << 16));
    g1[3] = (int)(((tensor_d1 >> 16) & 0xFFFFu) | ((tile_d0 & 0xFFFFu) << 16));
    g1[4] = (int)(tile_d1 & 0xFFFFu);                        // tile_dim1, tile_dim2=0
    g1[5] = (int)stride;                                     // tensor_dim0_stride lo
    g1[6] = 0;
    g1[7] = 0;
    const i32x4 z4 = {0, 0, 0, 0};
    const i32x8 z8 = {0, 0, 0, 0, 0, 0, 0, 0};
    __builtin_amdgcn_tensor_load_to_lds(g0, g1, z4, z4, z8, 0);
}

// ---------------------------------------------------------------------------
// Projection GEMM: out = W[O x CIN] @ X[CIN x N] + bias, stored bf16.
// TRANS=1 -> out[b, n, o] (qT / kT); TRANS=0 -> out[b, o, n] (v).
// Block: 128 threads = 4 waves; block tile = 64(o) x 16(n).
// ---------------------------------------------------------------------------
template <int O, int CIN, bool TRANS>
__global__ __launch_bounds__(128) void proj_gemm(
    const float* __restrict__ Wmat, const float* __restrict__ bias,
    const float* __restrict__ X, __bf16* __restrict__ out)
{
    // X tile staged TRANSPOSED: [16 n][40 c] (pad -> 80B row, 16B aligned reads)
    __shared__ __bf16 xTile[16 * 40];

    const int b     = blockIdx.z;
    const int nBase = blockIdx.x * 16;
    const int wave  = threadIdx.x >> 5;
    const int lane  = threadIdx.x & 31;
    const int h     = lane >> 4;
    const int l15   = lane & 15;
    const int oBase = blockIdx.y * 64 + wave * 16;

    const float* __restrict__ Xb = X + (size_t)b * CIN * N_;

    f32x8 acc = {};                            // D tile [16 o x 16 n]

    for (int c0 = 0; c0 < CIN; c0 += 32) {
        // --- cooperative transpose-stage of X tile [32 c x 16 n] -> LDS [n][c]
        {
            const int row = threadIdx.x >> 2;          // c within tile: 0..31
            const int nq  = (threadIdx.x & 3) * 4;     // n within tile: 0,4,8,12
            f32x4 v4 = *(const f32x4*)(Xb + (size_t)(c0 + row) * N_ + nBase + nq);
            xTile[(nq + 0) * 40 + row] = (__bf16)v4.x;
            xTile[(nq + 1) * 40 + row] = (__bf16)v4.y;
            xTile[(nq + 2) * 40 + row] = (__bf16)v4.z;
            xTile[(nq + 3) * 40 + row] = (__bf16)v4.w;
        }
        __syncthreads();

        // --- A fragment: W rows o, 32 c.  e<8: c=h*8+e ; e>=8: c=h*8+16+(e-8)
        bf16x16 aF;
        {
            const float* wr = Wmat + (size_t)(oBase + l15) * CIN + c0 + h * 8;
            #pragma unroll
            for (int e = 0; e < 8; ++e) aF[e]     = (__bf16)wr[e];
            #pragma unroll
            for (int e = 0; e < 8; ++e) aF[8 + e] = (__bf16)wr[16 + e];
        }
        // --- B fragment: n = l15; e<8: c=h*8+e ; e>=8: c=h*8+16+(e-8)
        bf16x16 bF = cat8(*(const bf16x8*)&xTile[l15 * 40 + h * 8], *(const bf16x8*)&xTile[l15 * 40 + h * 8 + 16]);

        acc = wmma_bf16(aF, bF, acc);
        __syncthreads();
    }

    // --- epilogue: +bias, bf16 store (D: row o = r + 8h, col n = l15)
    #pragma unroll
    for (int r = 0; r < 8; ++r) {
        const int o = oBase + r + 8 * h;
        const int n = nBase + l15;
        const float val = acc[r] + bias[o];
        if (TRANS) out[((size_t)b * N_ + n) * O + o] = (__bf16)val;
        else       out[((size_t)b * O  + o) * N_ + n] = (__bf16)val;
    }
}

// ---------------------------------------------------------------------------
// Flash attention + output projection + residual.
// Block: 256 threads = 8 waves = 4 n-tiles x 2 o-halves.
// Per 32-m step: wave0 TDM-loads K tile [32 m x 128 c] and V tile
// [256 o x 32 m] into LDS (shared by all 8 waves), waits TENSORcnt, barrier.
// ---------------------------------------------------------------------------
__global__ __launch_bounds__(256) void flash_attn(
    const __bf16* __restrict__ qT,     // [B, N, CI]
    const __bf16* __restrict__ kT,     // [B, M, CI]
    const __bf16* __restrict__ vv,     // [B, C, M]
    const float*  __restrict__ x_main, // [B, C, N]
    const float*  __restrict__ gammap,
    float* __restrict__ outp)          // [B, C, N]
{
    __shared__ __bf16 kTile[32 * 128];         // [m'][c]   8 KB
    __shared__ __bf16 vTile[256 * 32];         // [o][m']  16 KB
    __shared__ __bf16 pTile[8][16 * 32];       // per-wave P relayout  8 KB

    const int b     = blockIdx.y;
    const int wave  = threadIdx.x >> 5;
    const int lane  = threadIdx.x & 31;
    const int h     = lane >> 4;
    const int l15   = lane & 15;
    const int nBase = (blockIdx.x * 4 + (wave >> 1)) * 16;
    const int oHalf = wave & 1;

    const __bf16* __restrict__ qTb = qT + ((size_t)b * N_ + nBase) * CI_;
    const __bf16* __restrict__ kTb = kT + (size_t)b * N_ * CI_;
    const __bf16* __restrict__ vvb = vv + (size_t)b * C_ * N_;

    const unsigned kTileAddr = (unsigned)(uintptr_t)(void*)kTile;
    const unsigned vTileAddr = (unsigned)(uintptr_t)(void*)vTile;

    // ---- resident Q A-fragments: [16 n x 32 c], n = l15
    bf16x16 qF[4];
    #pragma unroll
    for (int cc = 0; cc < 4; ++cc) {
        const __bf16* qr = qTb + (size_t)l15 * CI_ + cc * 32 + h * 16;
        qF[cc] = cat8(*(const bf16x8*)qr, *(const bf16x8*)(qr + 8));
    }

    f32x8 acc[8];
    #pragma unroll
    for (int ch = 0; ch < 8; ++ch) acc[ch] = f32x8{};
    float rowMax[8], rowSum[8];
    #pragma unroll
    for (int r = 0; r < 8; ++r) { rowMax[r] = -3.0e38f; rowSum[r] = 0.0f; }

    for (int m0 = 0; m0 < N_; m0 += 32) {
        // ---- async tensor loads of shared K / V tiles (wave 0 issues)
        if (wave == 0) {
            tdm_load_2d_bf16(kTileAddr, kTb + (size_t)m0 * CI_,
                             /*tile*/ CI_, 32, /*tensor*/ CI_, N_, /*stride*/ CI_);
            tdm_load_2d_bf16(vTileAddr, vvb + m0,
                             /*tile*/ 32, C_, /*tensor*/ N_, C_, /*stride*/ N_);
            __builtin_amdgcn_s_wait_tensorcnt(0);
        }
        __syncthreads();

        // ---- E tiles: two 16x16 (m and m+16); K B-frags from LDS
        f32x8 e0 = {}, e1 = {};
        #pragma unroll
        for (int cc = 0; cc < 4; ++cc) {
            bf16x16 k0 = *(const bf16x16*)&kTile[(l15)      * 128 + cc * 32 + h * 16];
            bf16x16 k1 = *(const bf16x16*)&kTile[(16 + l15) * 128 + cc * 32 + h * 16];
            e0 = wmma_bf16(qF[cc], k0, e0);
            e1 = wmma_bf16(qF[cc], k1, e1);
        }

        // ---- online softmax (row n = r + 8h lives in 16 lanes of one VGPR)
        float scale[8];
        #pragma unroll
        for (int r = 0; r < 8; ++r) {
            float t = fmaxf(e0[r], e1[r]);
            t = fmaxf(t, __shfl_xor(t, 1, 16));
            t = fmaxf(t, __shfl_xor(t, 2, 16));
            t = fmaxf(t, __shfl_xor(t, 4, 16));
            t = fmaxf(t, __shfl_xor(t, 8, 16));
            const float nm = fmaxf(rowMax[r], t);
            scale[r]  = __expf(rowMax[r] - nm);
            rowMax[r] = nm;
            const float p0 = __expf(e0[r] - nm);
            const float p1 = __expf(e1[r] - nm);
            float s = p0 + p1;
            s += __shfl_xor(s, 1, 16);
            s += __shfl_xor(s, 2, 16);
            s += __shfl_xor(s, 4, 16);
            s += __shfl_xor(s, 8, 16);
            rowSum[r] = rowSum[r] * scale[r] + s;
            __bf16* pd = &pTile[wave][(r + 8 * h) * 32];
            pd[l15]      = (__bf16)p0;
            pd[16 + l15] = (__bf16)p1;
        }

        // ---- rescale running output accumulators
        #pragma unroll
        for (int ch = 0; ch < 8; ++ch)
            #pragma unroll
            for (int r = 0; r < 8; ++r)
                acc[ch][r] *= scale[r];

        // ---- P A-fragment from LDS: n = l15; m = h*16 + e
        const __bf16* pr = &pTile[wave][l15 * 32 + h * 16];
        bf16x16 pF = cat8(*(const bf16x8*)pr, *(const bf16x8*)(pr + 8));

        // ---- P @ V^T : Vt B-frag from LDS, row m' = h*16+e contiguous
        #pragma unroll
        for (int ch = 0; ch < 8; ++ch) {
            const int o = oHalf * 128 + ch * 16 + l15;
            bf16x16 vF = *(const bf16x16*)&vTile[o * 32 + h * 16];
            acc[ch] = wmma_bf16(pF, vF, acc[ch]);
        }
        __syncthreads();   // protect kTile/vTile before next TDM overwrite
    }

    // ---- epilogue: normalize, gamma * O + x_main  (row n = r + 8h, col o = l15)
    const float g = gammap[0];
    #pragma unroll
    for (int ch = 0; ch < 8; ++ch) {
        const int o = oHalf * 128 + ch * 16 + l15;
        #pragma unroll
        for (int r = 0; r < 8; ++r) {
            const int n = nBase + r + 8 * h;
            const size_t idx = ((size_t)b * C_ + o) * N_ + n;
            outp[idx] = g * (acc[ch][r] / rowSum[r]) + x_main[idx];
        }
    }
}

// ---------------------------------------------------------------------------
extern "C" void kernel_launch(void* const* d_in, const int* in_sizes, int n_in,
                              void* d_out, int out_size, void* d_ws, size_t ws_size,
                              hipStream_t stream) {
    const float* x_main = (const float*)d_in[0];
    const float* z_p    = (const float*)d_in[1];
    const float* Wq     = (const float*)d_in[2];
    const float* bq     = (const float*)d_in[3];
    const float* Wk     = (const float*)d_in[4];
    const float* bk     = (const float*)d_in[5];
    const float* Wv     = (const float*)d_in[6];
    const float* bv     = (const float*)d_in[7];
    const float* gamma  = (const float*)d_in[8];
    float* outp = (float*)d_out;

    __bf16* qT = (__bf16*)d_ws;
    __bf16* kT = qT + (size_t)B_ * N_ * CI_;
    __bf16* vv = kT + (size_t)B_ * N_ * CI_;

    proj_gemm<CI_, C_,  true ><<<dim3(N_ / 16, CI_ / 64, B_), 128, 0, stream>>>(Wq, bq, x_main, qT);
    proj_gemm<CI_, Cs_, true ><<<dim3(N_ / 16, CI_ / 64, B_), 128, 0, stream>>>(Wk, bk, z_p,    kT);
    proj_gemm<C_,  Cs_, false><<<dim3(N_ / 16, C_  / 64, B_), 128, 0, stream>>>(Wv, bv, z_p,    vv);

    flash_attn<<<dim3(N_ / 64, B_), 256, 0, stream>>>(qT, kT, vv, x_main, gamma, outp);
}
